// ParserLSTM_25769803928
// MI455X (gfx1250) — matrix operand order repaired and checked by reading, not compiled, not run on hardware
//
#include <hip/hip_runtime.h>

// ---------------------------------------------------------------------------
// Bidirectional 3-layer LSTM for MI455X (gfx1250, wave32, WMMA bf16)
// B=64, T=512, D=200, H=400, 4H=1600
// ---------------------------------------------------------------------------

typedef __bf16 bf16_t;
typedef __attribute__((ext_vector_type(16))) __bf16 v16bf;
typedef __attribute__((ext_vector_type(8)))  __bf16 v8bf;
typedef __attribute__((ext_vector_type(8)))  float  v8f;
typedef __attribute__((ext_vector_type(4)))  int    v4i;

typedef __attribute__((address_space(1))) v4i* gv4i_p;
typedef __attribute__((address_space(3))) v4i* lv4i_p;

#define BATCH   64
#define TLEN    512
#define HID     400
#define HP      416          // 13*32 padded K for recurrent GEMM
#define KS_H    13
#define G4      1600         // 4H
#define NTILES  100          // 1600/16 column tiles
#define NBLK    25           // hidden-unit tile blocks per direction (400/16)
#define MTILES  2048         // (T*B)/16 row tiles

#ifndef __has_builtin
#define __has_builtin(x) 0
#endif

#if __has_builtin(__builtin_amdgcn_global_load_async_to_lds_b128)
#define HAVE_ASYNC_LDS 1
#else
#define HAVE_ASYNC_LDS 0
#endif

// Copy 16 bf16 (32B as 2x16B) global -> LDS, async if available.
static __device__ __forceinline__ void g2l_16(const bf16_t* g, bf16_t* l) {
#if HAVE_ASYNC_LDS
    __builtin_amdgcn_global_load_async_to_lds_b128((gv4i_p)g, (lv4i_p)l, 0, 0);
    __builtin_amdgcn_global_load_async_to_lds_b128((gv4i_p)(g + 8), (lv4i_p)(l + 8), 0, 0);
#else
    *(v8bf*)l       = *(const v8bf*)g;
    *(v8bf*)(l + 8) = *(const v8bf*)(g + 8);
#endif
}

static __device__ __forceinline__ void g2l_wait() {
#if HAVE_ASYNC_LDS
#if __has_builtin(__builtin_amdgcn_s_wait_asynccnt)
    __builtin_amdgcn_s_wait_asynccnt(0);
#else
    asm volatile("s_wait_asynccnt 0x0" ::: "memory");
#endif
#endif
}

static __device__ __forceinline__ v8f wmma_bf16(v16bf a, v16bf b, v8f c) {
    return __builtin_amdgcn_wmma_f32_16x16x32_bf16(
        /*neg_a=*/false, a, /*neg_b=*/false, b,
        /*c_mod=*/(short)0, c, /*reuse_a=*/false, /*reuse_b=*/false);
}

// A fragment (16x32, bf16): lane L holds row (L%16); K range
// [k0 .. k0+15] with k0 = (L<16 ? 0 : 16).
static __device__ __forceinline__ v16bf make_afrag(const bf16_t* __restrict__ row,
                                                   int kbase, int lane) {
    const int c0 = (lane < 16) ? 0 : 16;
    v8bf lo = *(const v8bf*)(row + kbase + c0);
    v8bf hi = *(const v8bf*)(row + kbase + c0 + 8);
    v16bf a;
#pragma unroll
    for (int i = 0; i < 8; ++i) { a[i] = lo[i]; a[i + 8] = hi[i]; }
    return a;
}

// Fast device transcendentals (v_exp_f32 + v_rcp_f32)
static __device__ __forceinline__ float sigm(float x) {
    return __builtin_amdgcn_rcpf(1.0f + __expf(-x));
}
static __device__ __forceinline__ float tanh_fast(float x) {
    x = fminf(fmaxf(x, -15.0f), 15.0f);
    float e = __expf(2.0f * x);
    return (e - 1.0f) * __builtin_amdgcn_rcpf(e + 1.0f);
}

// ---------------------------------------------------------------------------
// Packing kernels
// ---------------------------------------------------------------------------

// Pack weight matrix W [1600][Kreal] (two dir copies, stride dirStrideW) into
// B-fragment layout dst[2][100][KS][lane=32][16 halves], zero-padded K.
// lane L, half h: N = L%16, K = (L<16?0:16)+h;  B(k,n) = W[n][k].
__global__ void pack_b_kernel(const float* __restrict__ W, bf16_t* __restrict__ dst,
                              int Kreal, int KS, size_t dirStrideW) {
    size_t idx = (size_t)blockIdx.x * blockDim.x + threadIdx.x;
    size_t total = (size_t)2 * NTILES * KS * 32;
    if (idx >= total) return;
    int lane = (int)(idx & 31); size_t r = idx >> 5;
    int ks = (int)(r % KS); r /= KS;
    int nt = (int)(r % NTILES); int dir = (int)(r / NTILES);
    const float* Wd = W + (size_t)dir * dirStrideW;
    const int n = nt * 16 + (lane & 15);
    const int kbase = ks * 32 + ((lane < 16) ? 0 : 16);
    bf16_t* out = dst + idx * 16;
#pragma unroll
    for (int h = 0; h < 16; ++h) {
        int k = kbase + h;
        float v = (k < Kreal) ? Wd[(size_t)n * Kreal + k] : 0.0f;
        out[h] = (bf16_t)v;
    }
}

// x [B][T][D] f32 -> act0 [T*B][224] bf16 (time-major rows = t*64+b, K padded)
__global__ void pack_x_kernel(const float* __restrict__ x, bf16_t* __restrict__ act0) {
    size_t idx = (size_t)blockIdx.x * blockDim.x + threadIdx.x;
    const size_t total = (size_t)TLEN * BATCH * 224;
    if (idx >= total) return;
    int k = (int)(idx % 224);
    size_t row = idx / 224;
    int b = (int)(row % BATCH);
    int t = (int)(row / BATCH);
    float v = (k < 200) ? x[((size_t)b * TLEN + t) * 200 + k] : 0.0f;
    act0[idx] = (bf16_t)v;
}

__global__ void bias_kernel(const float* __restrict__ bih, const float* __restrict__ bhh,
                            float* __restrict__ dst) {
    int i = blockIdx.x * blockDim.x + threadIdx.x;
    if (i < 3 * 2 * G4) dst[i] = bih[i] + bhh[i];
}

__global__ void zero_state_kernel(bf16_t* __restrict__ hbuf, unsigned int* __restrict__ bar) {
    size_t i = (size_t)blockIdx.x * blockDim.x + threadIdx.x;
    if (i < (size_t)2 * 2 * BATCH * HP) hbuf[i] = (bf16_t)0.0f;
    if (i < 2) bar[i] = 0u;
}

// ---------------------------------------------------------------------------
// Input projection GEMM: xW = act @ Wih^T + bias, dumped as raw C fragments.
// Templated on KSI so the pipelined K loop fully unrolls (no register-rotation
// copies). Block = 8 waves sharing one ntile-quad; B panel staged in dynamic
// LDS once (async-to-LDS); each wave owns one 16-row mtile.
// xwfrag layout: [dir][mtile][ntile][lane][8] bf16
// ---------------------------------------------------------------------------
template <int KSI>
__global__ __launch_bounds__(256) void input_gemm_kernel(
    const bf16_t* __restrict__ act,   // [32768][Kp]
    const bf16_t* __restrict__ wB,    // [2][100][KSI][32][16]
    const float*  __restrict__ bias,  // [2][1600]
    bf16_t* __restrict__ xwfrag,      // [2][2048][100][32][8]
    int Kp) {
    extern __shared__ bf16_t ldsB[];  // [KSI][4][32][16] halves
    const int tid  = threadIdx.x;
    const int wave = tid >> 5;
    const int lane = tid & 31;

    int task = blockIdx.x;                    // 2 * 256 * 25 blocks
    const int nq  = task % NBLK; task /= NBLK;
    const int mb  = task % (MTILES / 8);
    const int dir = task / (MTILES / 8);
    const int nrow = lane & 15;

    // --- stage B panel into LDS (32B units: KSI*4*32) ---
    const int nUnits = KSI * 128;
    for (int u = tid; u < nUnits; u += 256) {
        const int l  = u & 31;
        const int g  = (u >> 5) & 3;
        const int ks = u >> 7;
        const bf16_t* src = wB + (((size_t)dir * NTILES + nq * 4 + g) * KSI + ks) * 512
                               + (size_t)l * 16;
        bf16_t* dst = ldsB + ((size_t)(ks * 4 + g) * 32 + l) * 16;
        g2l_16(src, dst);
    }
    g2l_wait();
    __syncthreads();

    // --- accumulators init with bias ---
    v8f acc[4];
#pragma unroll
    for (int g = 0; g < 4; ++g) {
        int nt = nq * 4 + g;
        float bb = bias[(size_t)dir * G4 + nt * 16 + nrow];
#pragma unroll
        for (int i = 0; i < 8; ++i) acc[g][i] = bb;
    }

    const int mt = mb * 8 + wave;
    const bf16_t* arow = act + (size_t)(mt * 16 + nrow) * Kp;

    // Software-pipelined, fully unrolled K loop (SSA kills the rotation)
    v16bf af = make_afrag(arow, 0, lane);
    v16bf bcur[4];
#pragma unroll
    for (int g = 0; g < 4; ++g)
        bcur[g] = *(const v16bf*)(ldsB + ((size_t)g * 32 + lane) * 16);

#pragma unroll
    for (int ks = 0; ks < KSI; ++ks) {
        v16bf afn;
        v16bf bn[4];
        if (ks + 1 < KSI) {
            afn = make_afrag(arow, (ks + 1) * 32, lane);
#pragma unroll
            for (int g = 0; g < 4; ++g)
                bn[g] = *(const v16bf*)(ldsB + ((size_t)((ks + 1) * 4 + g) * 32 + lane) * 16);
        }
#pragma unroll
        for (int g = 0; g < 4; ++g)
            acc[g] = wmma_bf16(af, bcur[g], acc[g]);
        af = afn;
#pragma unroll
        for (int g = 0; g < 4; ++g) bcur[g] = bn[g];
    }

#pragma unroll
    for (int g = 0; g < 4; ++g) {
        int nt = nq * 4 + g;
        bf16_t* dst = xwfrag + ((((size_t)dir * MTILES + mt) * NTILES + nt) * 32 + lane) * 8;
        v8bf o;
#pragma unroll
        for (int i = 0; i < 8; ++i) o[i] = (bf16_t)acc[g][i];
        *(v8bf*)dst = o;
    }
}

// ---------------------------------------------------------------------------
// Persistent recurrent scan: 50 blocks (2 dirs x 25 hidden tiles), 128 thr.
// Whh B-fragments (time-invariant, 53KB) preloaded to LDS once; c in regs;
// h double-buffered bf16 in global; atomic barrier per timestep per dir;
// K loop software-pipelined & fully unrolled; next xW fragments prefetched.
// ---------------------------------------------------------------------------
template <bool IS_LAST>
__global__ __launch_bounds__(128) void lstm_scan_kernel(
    const bf16_t* __restrict__ xwfrag,  // [2][2048][100][32][8]
    const bf16_t* __restrict__ whhB,    // [2][100][13][32][16]
    bf16_t* __restrict__ hbuf,          // [2][2][64][416]
    unsigned int* __restrict__ bar,     // [2]
    const float* __restrict__ mask,     // [64][512]
    bf16_t* __restrict__ actOut,        // [T][B][800] bf16 (layers 0,1)
    float* __restrict__ finalOut) {     // [B][T][800] f32 (layer 2)
    __shared__ bf16_t whhLds[52 * 512];     // [ks=13][g=4][lane=32][16] = 53248 B

    const int dir  = blockIdx.x / NBLK;
    const int jb   = blockIdx.x % NBLK;
    const int u0   = jb * 16;
    const int tid  = threadIdx.x;
    const int wave = tid >> 5;
    const int lane = tid & 31;
    const int m0   = wave * 16;
    const int nrow = lane & 15;
    const int rowoff = (lane < 16) ? 0 : 8;

    bf16_t* hdir = hbuf + (size_t)dir * (2 * BATCH * HP);
    const bf16_t* whhD = whhB + (size_t)dir * NTILES * KS_H * 512;

    // --- preload Whh fragments for this block's 4 gate ntiles into LDS ---
    for (int u = tid; u < 52 * 32; u += 128) {     // 32B units
        const int l = u & 31;
        const int frag = u >> 5;
        const int ks = frag >> 2;
        const int g  = frag & 3;
        const int nt = g * NBLK + jb;
        const bf16_t* src = whhD + ((size_t)nt * KS_H + ks) * 512 + (size_t)l * 16;
        bf16_t* dst = whhLds + ((size_t)frag * 32 + l) * 16;
        g2l_16(src, dst);
    }
    g2l_wait();
    __syncthreads();

    v8f c = {};
    unsigned int target = 0;

    for (int t = 0; t < TLEN; ++t) {
        const int tt = dir ? (TLEN - 1 - t) : t;
        const bf16_t* hin  = hdir + (size_t)(t & 1) * (BATCH * HP);
        bf16_t*       hout = hdir + (size_t)((t + 1) & 1) * (BATCH * HP);

        // Init accumulators from precomputed xW fragments (identical C layout)
        const int mtile = tt * 4 + wave;
        v8f acc[4];
#pragma unroll
        for (int g = 0; g < 4; ++g) {
            const int nt = g * NBLK + jb;
            const bf16_t* p = xwfrag +
                ((((size_t)dir * MTILES + mtile) * NTILES + nt) * 32 + lane) * 8;
            v8bf xv = *(const v8bf*)p;
            v8f a;
#pragma unroll
            for (int i = 0; i < 8; ++i) a[i] = (float)xv[i];
            acc[g] = a;
        }

        // Recurrent GEMM: gates += h_prev @ Whh^T, B from LDS, pipelined
        const bf16_t* arow = hin + (size_t)(m0 + nrow) * HP;
        v16bf af = make_afrag(arow, 0, lane);
        v16bf bcur[4];
#pragma unroll
        for (int g = 0; g < 4; ++g)
            bcur[g] = *(const v16bf*)(whhLds + ((size_t)g * 32 + lane) * 16);

#pragma unroll
        for (int ks = 0; ks < KS_H; ++ks) {
            v16bf afn;
            v16bf bn[4];
            if (ks + 1 < KS_H) {
                afn = make_afrag(arow, (ks + 1) * 32, lane);
#pragma unroll
                for (int g = 0; g < 4; ++g)
                    bn[g] = *(const v16bf*)(whhLds +
                                ((size_t)((ks + 1) * 4 + g) * 32 + lane) * 16);
            }
#pragma unroll
            for (int g = 0; g < 4; ++g)
                acc[g] = wmma_bf16(af, bcur[g], acc[g]);
            af = afn;
#pragma unroll
            for (int g = 0; g < 4; ++g) bcur[g] = bn[g];
        }

        // Prefetch next timestep's xW fragments while we finish this step
        if (t + 1 < TLEN) {
            const int ttn = dir ? (TLEN - 2 - t) : (t + 1);
            const int mtn = ttn * 4 + wave;
#pragma unroll
            for (int g = 0; g < 4; ++g) {
                const int nt = g * NBLK + jb;
                __builtin_prefetch(
                    xwfrag + ((((size_t)dir * MTILES + mtn) * NTILES + nt) * 32 + lane) * 8,
                    0, 1);
            }
        }

        // LSTM cell elementwise + state update + stores
        const int ucol = u0 + nrow;
#pragma unroll
        for (int e = 0; e < 8; ++e) {
            const int b = m0 + e + rowoff;
            float i_ = sigm(acc[0][e]);
            float f_ = sigm(acc[1][e]);
            float g_ = tanh_fast(acc[2][e]);
            float o_ = sigm(acc[3][e]);
            float c2 = f_ * c[e] + i_ * g_;
            float h2 = o_ * tanh_fast(c2);
            float mt = mask[(size_t)b * TLEN + tt];
            h2 *= mt; c2 *= mt;
            c[e] = c2;
            hout[(size_t)b * HP + ucol] = (bf16_t)h2;
            if (IS_LAST) {
                finalOut[((size_t)b * TLEN + tt) * (2 * HID) + dir * HID + ucol] = h2;
            } else {
                actOut[((size_t)tt * BATCH + b) * (2 * HID) + dir * HID + ucol] = (bf16_t)h2;
            }
        }

        // Grid barrier across the 25 blocks of this direction
        __threadfence();
        __syncthreads();
        target += NBLK;
        if (tid == 0) {
            __hip_atomic_fetch_add(&bar[dir], 1u, __ATOMIC_RELEASE,
                                   __HIP_MEMORY_SCOPE_AGENT);
            while (__hip_atomic_load(&bar[dir], __ATOMIC_ACQUIRE,
                                     __HIP_MEMORY_SCOPE_AGENT) < target) {
                __builtin_amdgcn_s_sleep(2);
            }
        }
        __syncthreads();
    }
}

// ---------------------------------------------------------------------------
// Host launch
// ---------------------------------------------------------------------------
extern "C" void kernel_launch(void* const* d_in, const int* in_sizes, int n_in,
                              void* d_out, int out_size, void* d_ws, size_t ws_size,
                              hipStream_t stream) {
    const float* x     = (const float*)d_in[0];   // [64][512][200]
    const float* mask  = (const float*)d_in[1];   // [64][512]
    const float* Wih0  = (const float*)d_in[2];   // [2][1600][200]
    const float* Wih12 = (const float*)d_in[3];   // [2][2][1600][800]
    const float* Whh   = (const float*)d_in[4];   // [3][2][1600][400]
    const float* bih   = (const float*)d_in[5];   // [3][2][1600]
    const float* bhh   = (const float*)d_in[6];   // [3][2][1600]
    float* out = (float*)d_out;                   // [64][512][800]

    char* ws = (char*)d_ws;
    auto carve = [&](size_t bytes) -> char* {
        char* p = ws; ws += (bytes + 255) & ~(size_t)255; return p;
    };
    bf16_t* act0   = (bf16_t*)carve((size_t)TLEN * BATCH * 224 * 2);
    bf16_t* act1   = (bf16_t*)carve((size_t)TLEN * BATCH * 800 * 2);
    bf16_t* act2   = (bf16_t*)carve((size_t)TLEN * BATCH * 800 * 2);
    bf16_t* xwfrag = (bf16_t*)carve((size_t)2 * MTILES * NTILES * 256 * 2);
    bf16_t* wih0P  = (bf16_t*)carve((size_t)2 * NTILES * 7  * 512 * 2);
    bf16_t* wih1P  = (bf16_t*)carve((size_t)2 * NTILES * 25 * 512 * 2);
    bf16_t* wih2P  = (bf16_t*)carve((size_t)2 * NTILES * 25 * 512 * 2);
    bf16_t* whhP0  = (bf16_t*)carve((size_t)2 * NTILES * KS_H * 512 * 2);
    bf16_t* whhP1  = (bf16_t*)carve((size_t)2 * NTILES * KS_H * 512 * 2);
    bf16_t* whhP2  = (bf16_t*)carve((size_t)2 * NTILES * KS_H * 512 * 2);
    float*  biasC  = (float*) carve((size_t)3 * 2 * G4 * 4);
    bf16_t* hbuf   = (bf16_t*)carve((size_t)2 * 2 * BATCH * HP * 2);
    unsigned int* bar = (unsigned int*)carve(256);

    {
        size_t n = (size_t)2 * NTILES * 7 * 32;
        pack_b_kernel<<<(int)((n + 255) / 256), 256, 0, stream>>>(
            Wih0, wih0P, 200, 7, (size_t)1600 * 200);
    }
    {
        size_t n = (size_t)2 * NTILES * 25 * 32;
        pack_b_kernel<<<(int)((n + 255) / 256), 256, 0, stream>>>(
            Wih12, wih1P, 800, 25, (size_t)1600 * 800);
        pack_b_kernel<<<(int)((n + 255) / 256), 256, 0, stream>>>(
            Wih12 + (size_t)2 * 1600 * 800, wih2P, 800, 25, (size_t)1600 * 800);
    }
    {
        size_t n = (size_t)2 * NTILES * KS_H * 32;
        int blks = (int)((n + 255) / 256);
        pack_b_kernel<<<blks, 256, 0, stream>>>(Whh, whhP0, 400, KS_H, (size_t)1600 * 400);
        pack_b_kernel<<<blks, 256, 0, stream>>>(Whh + (size_t)2 * 1600 * 400, whhP1,
                                                400, KS_H, (size_t)1600 * 400);
        pack_b_kernel<<<blks, 256, 0, stream>>>(Whh + (size_t)4 * 1600 * 400, whhP2,
                                                400, KS_H, (size_t)1600 * 400);
    }
    bias_kernel<<<(3 * 2 * G4 + 255) / 256, 256, 0, stream>>>(bih, bhh, biasC);
    {
        size_t n = (size_t)TLEN * BATCH * 224;
        pack_x_kernel<<<(int)((n + 255) / 256), 256, 0, stream>>>(x, act0);
    }

    const bf16_t* whhP[3] = { whhP0, whhP1, whhP2 };

    const int gemmBlocks = 2 * (MTILES / 8) * NBLK;   // 12800
    const int zeroBlocks = (int)(((size_t)2 * 2 * BATCH * HP + 255) / 256);

    // ---- Layer 0 ----
    input_gemm_kernel<7><<<gemmBlocks, 256, (size_t)7 * 4096, stream>>>(
        act0, wih0P, biasC + 0 * 2 * G4, xwfrag, 224);
    zero_state_kernel<<<zeroBlocks, 256, 0, stream>>>(hbuf, bar);
    lstm_scan_kernel<false><<<2 * NBLK, 128, 0, stream>>>(
        xwfrag, whhP[0], hbuf, bar, mask, act1, out);

    // ---- Layer 1 ----
    input_gemm_kernel<25><<<gemmBlocks, 256, (size_t)25 * 4096, stream>>>(
        act1, wih1P, biasC + 1 * 2 * G4, xwfrag, 800);
    zero_state_kernel<<<zeroBlocks, 256, 0, stream>>>(hbuf, bar);
    lstm_scan_kernel<false><<<2 * NBLK, 128, 0, stream>>>(
        xwfrag, whhP[1], hbuf, bar, mask, act2, out);

    // ---- Layer 2 ----
    input_gemm_kernel<25><<<gemmBlocks, 256, (size_t)25 * 4096, stream>>>(
        act2, wih2P, biasC + 2 * 2 * G4, xwfrag, 800);
    zero_state_kernel<<<zeroBlocks, 256, 0, stream>>>(hbuf, bar);
    lstm_scan_kernel<true><<<2 * NBLK, 128, 0, stream>>>(
        xwfrag, whhP[2], hbuf, bar, mask, act2, out);
}
